// BigramLanguageModel_53841710023355
// MI455X (gfx1250) — matrix operand order repaired and checked by reading, not compiled, not run on hardware
//
#include <hip/hip_runtime.h>
#include <hip/hip_bf16.h>
#include <math.h>

typedef __attribute__((ext_vector_type(16))) __bf16 v16bf;
typedef __attribute__((ext_vector_type(8)))  __bf16 v8bf;
typedef __attribute__((ext_vector_type(8)))  float  v8f;

// ---------------------------------------------------------------------------
// gfx1250 async global->LDS copy (ASYNCcnt-tracked); fall back to a
// synchronous reg->LDS copy when the builtin is unavailable.
// Probe result (round 4): builtin exists, 4 args, param0 is
// 'int __attribute__((vector_size(16))) addrspace(1)*'.
// ---------------------------------------------------------------------------
#if defined(__has_builtin)
#if __has_builtin(__builtin_amdgcn_global_load_async_to_lds_b128)
#define HAVE_ASYNC_LDS 1
#endif
#if __has_builtin(__builtin_amdgcn_s_wait_asynccnt)
#define HAVE_WAIT_ASYNC_BUILTIN 1
#endif
#endif
#ifndef HAVE_ASYNC_LDS
#define HAVE_ASYNC_LDS 0
#endif
#ifndef HAVE_WAIT_ASYNC_BUILTIN
#define HAVE_WAIT_ASYNC_BUILTIN 0
#endif

typedef int v4i __attribute__((vector_size(16)));
typedef __attribute__((address_space(1))) v4i* gv4i_p;   // global 16B vector ptr
typedef __attribute__((address_space(3))) v4i* lv4i_p;   // LDS    16B vector ptr

static __device__ __forceinline__ void copy16_to_lds(const __bf16* g, __bf16* l) {
#if HAVE_ASYNC_LDS
  __builtin_amdgcn_global_load_async_to_lds_b128((gv4i_p)g, (lv4i_p)l, 0, 0);
#else
  *(v8bf*)l = *(const v8bf*)g;
#endif
}

static __device__ __forceinline__ void wait_async_all() {
#if HAVE_ASYNC_LDS
#if HAVE_WAIT_ASYNC_BUILTIN
  __builtin_amdgcn_s_wait_asynccnt(0);
#else
  asm volatile("s_wait_asynccnt 0" ::: "memory");
#endif
#endif
}

// ---------------------------------------------------------------------------
// WMMA fragment helpers (k order used by this code, wave32)
// A 16x32 bf16: lanes 0-15 row M=lane, K = 0..15 ; lanes 16-31 same
// rows, K = 16..31.
// B 32x16 bf16: lanes 0-15 col N=lane, K=0..15 ; lanes 16-31 col N=lane-16,
// K=16..31.
// C/D 16x16 f32: lane&15 = col, VGPR j = row j (+8 for lanes 16-31).
// ---------------------------------------------------------------------------

// 16 contiguous bf16 (B-frags from [N][K] operands): 2x b128
static __device__ __forceinline__ v16bf load_bf16x16(const __bf16* p) {
  v8bf lo = *(const v8bf*)p;
  v8bf hi = *(const v8bf*)(p + 8);
  return __builtin_shufflevector(lo, hi, 0, 1, 2, 3, 4, 5, 6, 7,
                                 8, 9, 10, 11, 12, 13, 14, 15);
}

// A-frag per-lane pattern: 16 contiguous at p
static __device__ __forceinline__ v16bf load_a_pair(const __bf16* p) {
  v8bf lo = *(const v8bf*)p;
  v8bf hi = *(const v8bf*)(p + 8);
  return __builtin_shufflevector(lo, hi, 0, 1, 2, 3, 4, 5, 6, 7,
                                 8, 9, 10, 11, 12, 13, 14, 15);
}

static __device__ __forceinline__ v16bf load_a_frag(const __bf16* A, int lda,
                                                    int m0, int k0, int lane) {
  return load_a_pair(A + (size_t)(m0 + (lane & 15)) * lda + k0 + ((lane >> 4) << 4));
}

static __device__ __forceinline__ v8f wmma_bf16(v16bf a, v16bf b, v8f c) {
  return __builtin_amdgcn_wmma_f32_16x16x32_bf16(false, a, false, b,
                                                 (short)0, c, false, false);
}

// Cross-half exchange (lane i <-> lane i^16) via v_permlanex16_b32 (VALU)
static __device__ __forceinline__ float xor16(float v) {
  int i = __builtin_bit_cast(int, v);
  int r = __builtin_amdgcn_permlanex16(i, i, 0x76543210, 0xfedcba98, false, false);
  return __builtin_bit_cast(float, r);
}

// ---------------------------------------------------------------------------
// bf16 GEMM: C[M,N] = alpha * A[M,K] @ B[K,N] (+bias)(+ReLU). B given
// TRANSPOSED as BT[N][K]. Block = 4 waves stacked on M, all sharing one
// 64-column B tile staged in LDS (double-buffered, async copy overlapped
// with WMMA). One wave -> 32x64 tile (2 A-frags x 4 B-frags per k-step).
// Optional: f32 accumulate-into (residual; regular stores), f32 streaming
// store (non-temporal), bf16 copy (row-major or TROUT column-major [N][M]).
// ---------------------------------------------------------------------------

// Stage 64(n) x 32(k) B tile (4KB) into LDS: 128 threads x 2 x 16B
static __device__ __forceinline__ void stage_b_tile(const __bf16* __restrict__ BT,
                                                    int K, int nblk, int k0,
                                                    __bf16* dst, int tid) {
#pragma unroll
  for (int i = 0; i < 2; ++i) {
    const int idx  = tid + (i << 7);   // 0..255 slots of 16B
    const int n    = idx >> 2;         // 0..63
    const int koff = (idx & 3) << 3;   // 0,8,16,24
    copy16_to_lds(BT + (size_t)(nblk + n) * K + k0 + koff, dst + n * 32 + koff);
  }
}

template <bool RELU, bool ACCUM, bool TROUT>
__global__ void __launch_bounds__(128)
gemm_bf16_kernel(const __bf16* __restrict__ A, const __bf16* __restrict__ BT,
                 const float* __restrict__ bias, float* __restrict__ outf,
                 __bf16* __restrict__ outb, int M, int N, int K, float alpha) {
  __shared__ __bf16 btile[2][64 * 32];   // 8KB double buffer
  const int tid  = threadIdx.x;
  const int lane = tid & 31;
  const int w    = tid >> 5;
  const int lc   = lane & 15;
  const int m0   = blockIdx.y * 128 + w * 32;
  const int nblk = blockIdx.x * 64;
  const int nk   = K >> 5;

  stage_b_tile(BT, K, nblk, 0, btile[0], tid);

  v8f acc[2][4] = {};
  for (int i = 0; i < nk; ++i) {
    const int k0 = i << 5;
    wait_async_all();
    __syncthreads();
    if (i + 1 < nk) {
      stage_b_tile(BT, K, nblk, k0 + 32, btile[(i + 1) & 1], tid);
      __builtin_prefetch((const void*)(A + (size_t)(m0 + lc) * K + k0 + 32), 0, 1);
    }
    v16bf a0 = load_a_frag(A, K, m0, k0, lane);
    v16bf a1 = load_a_frag(A, K, m0 + 16, k0, lane);
    const __bf16* bt = btile[i & 1] + ((lane >> 4) << 4);  // + half*16 along k
#pragma unroll
    for (int nt = 0; nt < 4; ++nt) {
      v16bf bf = load_bf16x16(bt + (nt * 16 + lc) * 32);
      acc[0][nt] = wmma_bf16(a0, bf, acc[0][nt]);
      acc[1][nt] = wmma_bf16(a1, bf, acc[1][nt]);
    }
  }

  const int cb = lane & 15;
#pragma unroll
  for (int mt = 0; mt < 2; ++mt) {
    const int rbase = m0 + mt * 16 + ((lane >> 4) << 3);
#pragma unroll
    for (int nt = 0; nt < 4; ++nt) {
      const int c = nblk + nt * 16 + cb;
      const float bv = bias ? bias[c] : 0.0f;
      float vv[8];
#pragma unroll
      for (int j = 0; j < 8; ++j) {
        float t = acc[mt][nt][j] * alpha + bv;
        if (RELU) t = fmaxf(t, 0.0f);
        vv[j] = t;
      }
      if (outf) {
#pragma unroll
        for (int j = 0; j < 8; ++j) {
          const size_t idx = (size_t)(rbase + j) * (size_t)N + (size_t)c;
          if (ACCUM) outf[idx] += vv[j];
          else       __builtin_nontemporal_store(vv[j], &outf[idx]);  // streaming logits
        }
      }
      if (outb) {
        if (TROUT) {  // [N][M]: 8 consecutive rows -> one b128 store
          v8bf o;
#pragma unroll
          for (int j = 0; j < 8; ++j) o[j] = (__bf16)vv[j];
          *(v8bf*)(outb + (size_t)c * (size_t)M + rbase) = o;
        } else {
#pragma unroll
          for (int j = 0; j < 8; ++j)
            outb[(size_t)(rbase + j) * (size_t)N + c] = (__bf16)vv[j];
        }
      }
    }
  }
}

// ---------------------------------------------------------------------------
// Flash attention, TRANSPOSED-score formulation. One wave per (b, head,
// 16-query tile); each lane owns one query column t = tlo + (lane&15).
//   S^T = K_tile(A, 16x32) @ Q^T(B, 32x16)  -> scores striped per-lane
//   att^T = V^T(A) @ P^T(B)                 -> accumulators per-lane
// Row max/sum are in-lane reductions + ONE cross-half permlanex16; the P^T
// B-frag is built from registers with 16 permlanex16 exchanges. No LDS.
// q comes pre-scaled by E^-0.5 (folded into Q projection); v is
// head-transposed vT[E][M].
// ---------------------------------------------------------------------------
__global__ void __launch_bounds__(128)
attn_kernel(const __bf16* __restrict__ q, const __bf16* __restrict__ k,
            const __bf16* __restrict__ vT, __bf16* __restrict__ att,
            int T, int E, int H, int Mtot) {
  const int lane = threadIdx.x & 31;
  const int wg   = blockIdx.x * 4 + (threadIdx.x >> 5);
  const int nqt  = T >> 4;
  const int qt   = wg % nqt;
  const int bh   = wg / nqt;
  const int b    = bh / H, hh = bh % H;
  const int rowbase = b * T;
  const int hoff    = hh * 32;
  const int half = lane >> 4;
  const int lc   = lane & 15;
  const int tlo  = qt * 16;
  const int tq   = tlo + lc;  // this lane's query position

  // Q^T B-frag: element (d, t) = q[t, d]; constant across the whole loop
  v16bf bq = load_bf16x16(q + (size_t)(rowbase + tq) * E + hoff + half * 16);

  v8f acc0 = {}, acc1 = {};
  float m = -INFINITY, l = 0.0f;

  for (int s0 = 0; s0 <= tlo; s0 += 32) {
    v16bf ak0 = load_a_frag(k + hoff, E, rowbase + s0, 0, lane);
    v8f z = {};
    v8f st0 = wmma_bf16(ak0, bq, z);
    v8f st1 = {};
    const bool have1 = (s0 + 16 <= tlo);
    if (have1) {
      v16bf ak1 = load_a_frag(k + hoff, E, rowbase + s0 + 16, 0, lane);
      st1 = wmma_bf16(ak1, bq, z);
    }

    float v0[8], v1[8];
#pragma unroll
    for (int j = 0; j < 8; ++j) {
      const int srow = s0 + j + 8 * half;
      v0[j] = (srow <= tq) ? st0[j] : -INFINITY;
      v1[j] = (have1 && (srow + 16 <= tq)) ? st1[j] : -INFINITY;
    }
    float tm = v0[0];
#pragma unroll
    for (int j = 1; j < 8; ++j) tm = fmaxf(tm, v0[j]);
#pragma unroll
    for (int j = 0; j < 8; ++j) tm = fmaxf(tm, v1[j]);
    tm = fmaxf(tm, xor16(tm));
    const float mnew = fmaxf(m, tm);
    const float corr = __expf(m - mnew);

    float p0[8], p1[8], ps = 0.0f;
#pragma unroll
    for (int j = 0; j < 8; ++j) { p0[j] = __expf(v0[j] - mnew); ps += p0[j]; }
#pragma unroll
    for (int j = 0; j < 8; ++j) { p1[j] = __expf(v1[j] - mnew); ps += p1[j]; }
    ps += xor16(ps);
    l = l * corr + ps;
    m = mnew;
#pragma unroll
    for (int j = 0; j < 8; ++j) { acc0[j] *= corr; acc1[j] *= corr; }

    // Build P^T B-frag (K = s0..s0+31, col t)
    v16bf pB;
#pragma unroll
    for (int j = 0; j < 8; ++j) {
      const float o0 = xor16(p0[j]);  // half0 <- partner p0 rows 8+j
      const float o1 = xor16(p1[j]);  // half1 <- partner p1 rows j
      const float lo = half ? o1 : p0[j];
      const float hi = half ? p1[j] : o0;
      pB[j]     = (__bf16)lo;
      pB[8 + j] = (__bf16)hi;
    }

    // att^T += V^T @ P^T ; V^T A-frags contiguous from vT[E][M]
    const __bf16* vp = vT + (size_t)(hoff + lc) * Mtot + rowbase + s0 + half * 16;
    v16bf aV0 = load_a_pair(vp);
    v16bf aV1 = load_a_pair(vp + (size_t)16 * Mtot);
    acc0 = wmma_bf16(aV0, pB, acc0);
    acc1 = wmma_bf16(aV1, pB, acc1);
  }

  const float inv = 1.0f / l;
  const size_t base = (size_t)(rowbase + tq) * E + hoff;
  v8bf o0, o1;
#pragma unroll
  for (int j = 0; j < 8; ++j) {
    o0[j] = (__bf16)(acc0[j] * inv);
    o1[j] = (__bf16)(acc1[j] * inv);
  }
  *(v8bf*)(att + base + 8 * half)      = o0;
  *(v8bf*)(att + base + 16 + 8 * half) = o1;
}

// ---------------------------------------------------------------------------
// LayerNorm over E=256 (one wave per row, 8 elems/lane), bf16 output
// ---------------------------------------------------------------------------
__global__ void __launch_bounds__(256)
ln_kernel(const float* __restrict__ x, const float* __restrict__ g,
          const float* __restrict__ bb, __bf16* __restrict__ out, int M, int E) {
  const int lane = threadIdx.x & 31;
  const int row  = blockIdx.x * 8 + (threadIdx.x >> 5);
  if (row >= M) return;
  const float* xr = x + (size_t)row * E;
  float vals[8];
  float s = 0.f, ss = 0.f;
#pragma unroll
  for (int i = 0; i < 8; ++i) {
    float vv = xr[lane * 8 + i];
    vals[i] = vv; s += vv; ss += vv * vv;
  }
#pragma unroll
  for (int off = 1; off < 32; off <<= 1) {
    s  += __shfl_xor(s, off, 32);
    ss += __shfl_xor(ss, off, 32);
  }
  const float mean = s / (float)E;
  const float var  = ss / (float)E - mean * mean;
  const float r    = rsqrtf(var + 1e-5f);
  __bf16* orow = out + (size_t)row * E;
#pragma unroll
  for (int i = 0; i < 8; ++i) {
    const int c = lane * 8 + i;
    orow[c] = (__bf16)((vals[i] - mean) * r * g[c] + bb[c]);
  }
}

// ---------------------------------------------------------------------------
__global__ void embed_kernel(const int* __restrict__ x, const float* __restrict__ tok,
                             const float* __restrict__ pos, float* __restrict__ h,
                             int T, int E, int total) {
  const int i = blockIdx.x * blockDim.x + threadIdx.x;
  if (i >= total) return;
  const int e  = i % E;
  const int bt = i / E;
  const int t  = bt % T;
  h[i] = tok[(size_t)x[bt] * E + e] + pos[(size_t)t * E + e];
}

// Tiled f32 [K][N] -> bf16 [N][K] transpose-convert (one-time weight prep)
__global__ void __launch_bounds__(256)
transpose_to_bf16_kernel(const float* __restrict__ in, __bf16* __restrict__ out,
                         int K, int N) {
  __shared__ float tile[32][33];
  const int kb = blockIdx.y * 32, nb = blockIdx.x * 32;
  const int tx = threadIdx.x & 31, ty = threadIdx.x >> 5;
  for (int r = ty; r < 32; r += 8)
    tile[r][tx] = in[(size_t)(kb + r) * N + nb + tx];
  __syncthreads();
  for (int r = ty; r < 32; r += 8)
    out[(size_t)(nb + r) * K + kb + tx] = (__bf16)tile[tx][r];
}

// ---------------------------------------------------------------------------
// Per-row single-pass online logsumexp loss (block per row, NT reads)
// ---------------------------------------------------------------------------
__global__ void __launch_bounds__(256)
row_loss_kernel(const float* __restrict__ logits, const int* __restrict__ y,
                float* __restrict__ rowloss, int V) {
  const int row = blockIdx.x;
  const int tid = threadIdx.x;
  const float* lr = logits + (size_t)row * V;
  float m = -INFINITY, s = 0.f;
  for (int c = tid; c < V; c += 256) {
    const float vv = __builtin_nontemporal_load(lr + c);
    if (vv > m) { s = s * __expf(m - vv) + 1.0f; m = vv; }
    else        { s += __expf(vv - m); }
  }
  __shared__ float rm[256], rs[256];
  rm[tid] = m; rs[tid] = s;
  __syncthreads();
  for (int st = 128; st > 0; st >>= 1) {
    if (tid < st) {
      const float m2 = rm[tid + st], s2 = rs[tid + st];
      const float Mx = fmaxf(rm[tid], m2);
      rs[tid] = rs[tid] * __expf(rm[tid] - Mx) + s2 * __expf(m2 - Mx);
      rm[tid] = Mx;
    }
    __syncthreads();
  }
  if (tid == 0) rowloss[row] = (rm[0] + __logf(rs[0])) - lr[y[row]];
}

__global__ void __launch_bounds__(256)
final_loss_kernel(const float* __restrict__ rowloss, float* __restrict__ out, int M) {
  __shared__ float red[256];
  const int tid = threadIdx.x;
  float s = 0.f;
  for (int i = tid; i < M; i += 256) s += rowloss[i];
  red[tid] = s;
  __syncthreads();
  for (int st = 128; st > 0; st >>= 1) {
    if (tid < st) red[tid] += red[tid + st];
    __syncthreads();
  }
  if (tid == 0) out[0] = red[0] / (float)M;
}

// ---------------------------------------------------------------------------
extern "C" void kernel_launch(void* const* d_in, const int* in_sizes, int n_in,
                              void* d_out, int out_size, void* d_ws, size_t ws_size,
                              hipStream_t stream) {
  (void)in_sizes; (void)n_in; (void)out_size; (void)ws_size;
  const int Bsz = 2, T = 2048, E = 256, H = 8, Lh = 6, V = 32000;
  const int M  = Bsz * T;   // 4096
  const int E4 = 4 * E;     // 1024

  const int*   x    = (const int*)d_in[0];
  const int*   y    = (const int*)d_in[1];
  const float* tok  = (const float*)d_in[2];
  const float* pos  = (const float*)d_in[3];
  const float* Wq   = (const float*)d_in[4];
  const float* Wk   = (const float*)d_in[5];
  const float* Wv   = (const float*)d_in[6];
  const float* Wp   = (const float*)d_in[7];
  const float* bp   = (const float*)d_in[8];
  const float* W1   = (const float*)d_in[9];
  const float* b1   = (const float*)d_in[10];
  const float* W2   = (const float*)d_in[11];
  const float* b2   = (const float*)d_in[12];
  const float* ln1g = (const float*)d_in[13];
  const float* ln1b = (const float*)d_in[14];
  const float* ln2g = (const float*)d_in[15];
  const float* ln2b = (const float*)d_in[16];
  const float* lnfg = (const float*)d_in[17];
  const float* lnfb = (const float*)d_in[18];
  const float* Wout = (const float*)d_in[19];
  const float* bout = (const float*)d_in[20];

  float* logits = (float*)d_out;
  float* lossp  = logits + (size_t)M * V;

  char* wsb = (char*)d_ws;
  size_t off = 0;
  auto alloc = [&](size_t bytes) -> void* {
    void* p = wsb + off;
    off += (bytes + 255) & ~(size_t)255;
    return p;
  };
  float*  h     = (float*) alloc((size_t)M * E * 4);
  __bf16* xnb   = (__bf16*)alloc((size_t)M * E * 2);
  __bf16* qb    = (__bf16*)alloc((size_t)M * E * 2);
  __bf16* kbuf  = (__bf16*)alloc((size_t)M * E * 2);
  __bf16* vtb   = (__bf16*)alloc((size_t)M * E * 2);   // head-transposed [E][M]
  __bf16* attb  = (__bf16*)alloc((size_t)M * E * 2);
  __bf16* hidb  = (__bf16*)alloc((size_t)M * E4 * 2);
  __bf16* wqb   = (__bf16*)alloc((size_t)Lh * E * E * 2);   // all weights: [N][K] bf16
  __bf16* wkb   = (__bf16*)alloc((size_t)Lh * E * E * 2);
  __bf16* wvb   = (__bf16*)alloc((size_t)Lh * E * E * 2);
  __bf16* wpb   = (__bf16*)alloc((size_t)Lh * E * E * 2);
  __bf16* w1b   = (__bf16*)alloc((size_t)Lh * E * E4 * 2);
  __bf16* w2b   = (__bf16*)alloc((size_t)Lh * E4 * E * 2);
  __bf16* woutb = (__bf16*)alloc((size_t)E * V * 2);
  float*  rowls = (float*) alloc((size_t)M * 4);

  auto convT = [&](const float* src, __bf16* dst, int K_, int N_) {
    transpose_to_bf16_kernel<<<dim3(N_ / 32, K_ / 32), 256, 0, stream>>>(src, dst, K_, N_);
  };
  for (int l = 0; l < Lh; ++l) {
    convT(Wq + (size_t)l * E * E,  wqb + (size_t)l * E * E,  E,  E);
    convT(Wk + (size_t)l * E * E,  wkb + (size_t)l * E * E,  E,  E);
    convT(Wv + (size_t)l * E * E,  wvb + (size_t)l * E * E,  E,  E);
    convT(Wp + (size_t)l * E * E,  wpb + (size_t)l * E * E,  E,  E);
    convT(W1 + (size_t)l * E * E4, w1b + (size_t)l * E * E4, E,  E4);
    convT(W2 + (size_t)l * E4 * E, w2b + (size_t)l * E4 * E, E4, E);
  }
  convT(Wout, woutb, E, V);

  embed_kernel<<<(M * E + 255) / 256, 256, 0, stream>>>(x, tok, pos, h, T, E, M * E);

  const float scale = 0.0625f;  // E^-0.5 = 1/16, folded into Q projection
  const dim3 gEE(E / 64, M / 128);
  const dim3 gE4(E4 / 64, M / 128);

  for (int l = 0; l < Lh; ++l) {
    ln_kernel<<<M / 8, 256, 0, stream>>>(h, ln1g + l * E, ln1b + l * E, xnb, M, E);
    gemm_bf16_kernel<false, false, false><<<gEE, 128, 0, stream>>>(
        xnb, wqb + (size_t)l * E * E, nullptr, nullptr, qb, M, E, E, scale);
    gemm_bf16_kernel<false, false, false><<<gEE, 128, 0, stream>>>(
        xnb, wkb + (size_t)l * E * E, nullptr, nullptr, kbuf, M, E, E, 1.0f);
    gemm_bf16_kernel<false, false, true><<<gEE, 128, 0, stream>>>(
        xnb, wvb + (size_t)l * E * E, nullptr, nullptr, vtb, M, E, E, 1.0f);
    attn_kernel<<<(Bsz * H * (T / 16)) / 4, 128, 0, stream>>>(
        qb, kbuf, vtb, attb, T, E, H, M);
    gemm_bf16_kernel<false, true, false><<<gEE, 128, 0, stream>>>(
        attb, wpb + (size_t)l * E * E, bp + l * E, h, nullptr, M, E, E, 1.0f);
    ln_kernel<<<M / 8, 256, 0, stream>>>(h, ln2g + l * E, ln2b + l * E, xnb, M, E);
    gemm_bf16_kernel<true, false, false><<<gE4, 128, 0, stream>>>(
        xnb, w1b + (size_t)l * E * E4, b1 + l * E4, nullptr, hidb, M, E4, E, 1.0f);
    gemm_bf16_kernel<false, true, false><<<gEE, 128, 0, stream>>>(
        hidb, w2b + (size_t)l * E4 * E, b2 + l * E, h, nullptr, M, E, E4, 1.0f);
  }

  ln_kernel<<<M / 8, 256, 0, stream>>>(h, lnfg, lnfb, xnb, M, E);
  const dim3 gV(V / 64, M / 128);
  gemm_bf16_kernel<false, false, false><<<gV, 128, 0, stream>>>(
      xnb, woutb, bout, logits, nullptr, M, V, E, 1.0f);

  row_loss_kernel<<<M, 256, 0, stream>>>(logits, y, rowls, V);
  final_loss_kernel<<<1, 256, 0, stream>>>(rowls, lossp, M);
}
